// _SparseAttnModule_29566554866379
// MI455X (gfx1250) — matrix operand order repaired and checked by reading, not compiled, not run on hardware
//
#include <hip/hip_runtime.h>

typedef __bf16 bf16_t;
typedef bf16_t v16bf __attribute__((ext_vector_type(16)));
typedef bf16_t v8bf  __attribute__((ext_vector_type(8)));
typedef float  v8f   __attribute__((ext_vector_type(8)));

#define NH      32      // heads
#define DD      128     // head dim
#define KTOT    256     // top-k keys per query
#define CHUNK   64      // keys per flash chunk
#define NCHUNK  4

#define QS_STRIDE 136   // 128 + 8 pad (bf16 elems)
#define KT_STRIDE 72    // 64  + 8 pad
#define VS_STRIDE 144   // 128 dims + 16 extra (ones column at 128) -> bank-rotating
#define PS_STRIDE 72

// log2-domain scaling: scores scaled by SCALE*log2(e) so every exp is a bare v_exp_f32
#define SCALE   0.08838834764831845f
#define LOG2E   1.4426950408889634f
#define SCALE2  (SCALE * LOG2E)

// ds_swizzle group-of-32 XOR patterns (offset = xor<<10 | or<<5 | and)
#define SWZ_X1 0x041F
#define SWZ_X2 0x081F
#define SWZ_X4 0x101F
#define SWZ_X8 0x201F

#define EXP2F(x) __builtin_amdgcn_exp2f(x)

static __device__ __forceinline__ v16bf cat16(v8bf lo, v8bf hi) {
    v16bf r;
#pragma unroll
    for (int i = 0; i < 8; ++i) { r[i] = lo[i]; r[i + 8] = hi[i]; }
    return r;
}

// raw v_max_num_f32: skips fmaxf's NaN-canonicalization self-max ops
static __device__ __forceinline__ float fmax_raw(float a, float b) {
    float d;
    asm("v_max_num_f32 %0, %1, %2" : "=v"(d) : "v"(a), "v"(b));
    return d;
}

static __device__ __forceinline__ float swz_maxstep(float v, const int imm) {
    float o;
    switch (imm) {   // imm must be a constant; fully unrolled callers only
        case SWZ_X1: o = __int_as_float(__builtin_amdgcn_ds_swizzle(__float_as_int(v), SWZ_X1)); break;
        case SWZ_X2: o = __int_as_float(__builtin_amdgcn_ds_swizzle(__float_as_int(v), SWZ_X2)); break;
        case SWZ_X4: o = __int_as_float(__builtin_amdgcn_ds_swizzle(__float_as_int(v), SWZ_X4)); break;
        default:     o = __int_as_float(__builtin_amdgcn_ds_swizzle(__float_as_int(v), SWZ_X8)); break;
    }
    return fmax_raw(v, o);
}

__global__ __launch_bounds__(256)
void sparse_attn_wmma_kernel(const float* __restrict__ q,
                             const float* __restrict__ kv,
                             const float* __restrict__ attn_sink,
                             const int*   __restrict__ topk,
                             float*       __restrict__ out)
{
    // LDS staging (bf16), padded strides to avoid bank conflicts
    __shared__ bf16_t qs[NH * QS_STRIDE];        // Q * SCALE2, row-major [head][dim]
    __shared__ bf16_t kt[DD * KT_STRIDE];        // K chunk, dim-major [dim][key]
    __shared__ bf16_t vs[CHUNK * VS_STRIDE];     // V chunk row-major + ones column at dim 128
    __shared__ bf16_t ps[NH * PS_STRIDE];        // exp2(S - m), row-major [head][key]
    __shared__ float  redm[4][NH];               // per-nTile partial row max
    __shared__ float  m_run[NH], m_new[NH], alpha_s[NH], sink_s[NH], lsum[NH], rinv[NH];

    const int s    = blockIdx.x;
    const int t    = threadIdx.x;
    const int lane = t & 31;
    const int w    = t >> 5;         // wave id 0..7
    const int mT   = w >> 2;         // head tile 0..1 (16 heads each)
    const int nt   = w & 3;          // key/dim sub-tile
    const int half = lane >> 4;      // 0: lanes 0-15, 1: lanes 16-31
    const int l15  = lane & 15;
    const int hb   = half * 8;       // A-fragment K sub-offset per ISA layout

    // ---- preload all chunk gather indices (breaks per-chunk index latency chain) ----
    const int gj  = t >> 2;          // key within chunk this thread gathers
    const int gq4 = t & 3;           // dim quarter
    const int gd0 = gq4 * 32;
    int idxs[NCHUNK];
#pragma unroll
    for (int c = 0; c < NCHUNK; ++c)
        idxs[c] = topk[(size_t)s * KTOT + c * CHUNK + gj];

    // ---- load Q (fold SCALE*log2e), convert to bf16, stage to LDS ----
    {
        const float4* qp = (const float4*)(q + (size_t)s * NH * DD);
        const int row = t >> 3;              // 8 threads per 128-wide row
        const int col = (t & 7) * 16;        // 16 floats per thread
        float4 f0 = qp[(row * DD + col) / 4 + 0];
        float4 f1 = qp[(row * DD + col) / 4 + 1];
        float4 f2 = qp[(row * DD + col) / 4 + 2];
        float4 f3 = qp[(row * DD + col) / 4 + 3];
        v8bf a = { (bf16_t)(f0.x*SCALE2), (bf16_t)(f0.y*SCALE2), (bf16_t)(f0.z*SCALE2), (bf16_t)(f0.w*SCALE2),
                   (bf16_t)(f1.x*SCALE2), (bf16_t)(f1.y*SCALE2), (bf16_t)(f1.z*SCALE2), (bf16_t)(f1.w*SCALE2) };
        v8bf b = { (bf16_t)(f2.x*SCALE2), (bf16_t)(f2.y*SCALE2), (bf16_t)(f2.z*SCALE2), (bf16_t)(f2.w*SCALE2),
                   (bf16_t)(f3.x*SCALE2), (bf16_t)(f3.y*SCALE2), (bf16_t)(f3.z*SCALE2), (bf16_t)(f3.w*SCALE2) };
        *(v8bf*)&qs[row * QS_STRIDE + col]     = a;
        *(v8bf*)&qs[row * QS_STRIDE + col + 8] = b;
    }
    // ones column for the fused row-sum GEMM: vs[key][128]=1, vs[key][129..143]=0 (constant)
    if (t < CHUNK) {
        v8bf one8  = { (bf16_t)1.0f, (bf16_t)0.0f, (bf16_t)0.0f, (bf16_t)0.0f,
                       (bf16_t)0.0f, (bf16_t)0.0f, (bf16_t)0.0f, (bf16_t)0.0f };
        v8bf zero8 = {};
        *(v8bf*)&vs[t * VS_STRIDE + 128] = one8;
        *(v8bf*)&vs[t * VS_STRIDE + 136] = zero8;
    }
    if (t < NH) {            // sink (log2 domain) acts as an extra "key" with no V contribution
        float sk = attn_sink[t] * LOG2E;
        sink_s[t] = sk;
        m_run[t]  = sk;
    }
    __syncthreads();

    // ---- hoist Q A-fragments into registers (loop-invariant across chunks) ----
    v16bf qa[4];
#pragma unroll
    for (int kb = 0; kb < 4; ++kb) {
        const int arow = (mT * 16 + l15) * QS_STRIDE + kb * 32;
        v8bf a0 = *(const v8bf*)&qs[arow + hb];
        v8bf a1 = *(const v8bf*)&qs[arow + hb + 16];
        qa[kb] = cat16(a0, a1);
    }

    v8f accA = {};           // out tile [16 heads x 16 dims], dt0
    v8f accB = {};           // dt1
    v8f accS = {};           // fused row-sum tile (only nt==0 waves use it)
    const int dt0 = nt * 2, dt1 = nt * 2 + 1;

#pragma unroll
    for (int c = 0; c < NCHUNK; ++c) {
        // ---- gather 64 keys: K -> kt (transposed), V -> vs (row-major) ----
        {
            const int idx = idxs[c];
            const float4* kp = (const float4*)(kv + (size_t)idx * 2 * DD + gd0);
            const float4* vp = (const float4*)(kv + (size_t)idx * 2 * DD + DD + gd0);
            // prefetch next chunk's K/V cache lines into L0/L2 while this clause drains
            if (c + 1 < NCHUNK) {
                const float* nk = kv + (size_t)idxs[c + 1] * 2 * DD + gd0;
                __builtin_prefetch(nk, 0, 3);
                __builtin_prefetch(nk + DD, 0, 3);
            }
#pragma unroll
            for (int i = 0; i < 8; ++i) {
                float4 f = kp[i];
                kt[(gd0 + i * 4 + 0) * KT_STRIDE + gj] = (bf16_t)f.x;
                kt[(gd0 + i * 4 + 1) * KT_STRIDE + gj] = (bf16_t)f.y;
                kt[(gd0 + i * 4 + 2) * KT_STRIDE + gj] = (bf16_t)f.z;
                kt[(gd0 + i * 4 + 3) * KT_STRIDE + gj] = (bf16_t)f.w;
            }
#pragma unroll
            for (int i = 0; i < 4; ++i) {
                float4 f0 = vp[i * 2 + 0];
                float4 f1 = vp[i * 2 + 1];
                v8bf pv = { (bf16_t)f0.x, (bf16_t)f0.y, (bf16_t)f0.z, (bf16_t)f0.w,
                            (bf16_t)f1.x, (bf16_t)f1.y, (bf16_t)f1.z, (bf16_t)f1.w };
                *(v8bf*)&vs[gj * VS_STRIDE + gd0 + i * 8] = pv;
            }
        }
        __syncthreads();

        // ---- S = Q @ K^T : two independent accumulator chains so fragment loads
        //      of one chain overlap the other chain's WMMA (avoids WAR lockstep) ----
        v8f sc0 = {}, sc1 = {};
#pragma unroll
        for (int kb = 0; kb < 2; ++kb) {
            const int dr0 = ((2 * kb + 0) * 32 + hb) * KT_STRIDE + nt * 16 + l15;
            const int dr1 = ((2 * kb + 1) * 32 + hb) * KT_STRIDE + nt * 16 + l15;
            v8bf e0, e1, f0, f1;
#pragma unroll
            for (int i = 0; i < 8; ++i) {
                e0[i] = kt[dr0 + i * KT_STRIDE];
                e1[i] = kt[dr0 + (16 + i) * KT_STRIDE];
                f0[i] = kt[dr1 + i * KT_STRIDE];
                f1[i] = kt[dr1 + (16 + i) * KT_STRIDE];
            }
            sc0 = __builtin_amdgcn_wmma_f32_16x16x32_bf16(false, qa[2 * kb + 0], false, cat16(e0, e1),
                                                          (short)0, sc0, false, false);
            sc1 = __builtin_amdgcn_wmma_f32_16x16x32_bf16(false, qa[2 * kb + 1], false, cat16(f0, f1),
                                                          (short)0, sc1, false, false);
        }
        v8f sc;
#pragma unroll
        for (int r = 0; r < 8; ++r) sc[r] = sc0[r] + sc1[r];

        // ---- per-wave row max via ds_swizzle XOR butterfly (halves independent) ----
        float pm[8];
#pragma unroll
        for (int r = 0; r < 8; ++r) {
            float v = sc[r];
            v = swz_maxstep(v, SWZ_X1);
            v = swz_maxstep(v, SWZ_X2);
            v = swz_maxstep(v, SWZ_X4);
            v = swz_maxstep(v, SWZ_X8);
            pm[r] = v;
        }
        if (l15 == 0) {
#pragma unroll
            for (int r = 0; r < 8; ++r) redm[nt][mT * 16 + half * 8 + r] = pm[r];
        }
        __syncthreads();

        // ---- online softmax state update (one thread per head) ----
        if (t < NH) {
            float mo = m_run[t];
            float mn = mo;
#pragma unroll
            for (int i = 0; i < 4; ++i) mn = fmax_raw(mn, redm[i][t]);
            m_run[t]   = mn;
            m_new[t]   = mn;
            alpha_s[t] = EXP2F(mo - mn);
        }
        __syncthreads();

        // ---- e = exp2(S - m), rescale accumulators, ps -> LDS (sums come from GEMM) ----
#pragma unroll
        for (int r = 0; r < 8; ++r) {
            const int row = mT * 16 + half * 8 + r;
            const float al = alpha_s[row];
            accA[r] *= al;
            accB[r] *= al;
            accS[r] *= al;
            float e = EXP2F(sc[r] - m_new[row]);
            ps[row * PS_STRIDE + nt * 16 + l15] = (bf16_t)e;
        }
        __syncthreads();

        // ---- O += P @ V : load all fragments per k-step, then issue wmmas ----
#pragma unroll
        for (int ks = 0; ks < 2; ++ks) {
            const int arow = (mT * 16 + l15) * PS_STRIDE + ks * 32;
            v8bf a0 = *(const v8bf*)&ps[arow + hb];
            v8bf a1 = *(const v8bf*)&ps[arow + hb + 16];
            const int krow = (ks * 32 + hb) * VS_STRIDE + l15;
            v8bf b0, b1, c0, c1;
#pragma unroll
            for (int i = 0; i < 8; ++i) {
                b0[i] = vs[krow + i * VS_STRIDE + dt0 * 16];
                b1[i] = vs[krow + (16 + i) * VS_STRIDE + dt0 * 16];
                c0[i] = vs[krow + i * VS_STRIDE + dt1 * 16];
                c1[i] = vs[krow + (16 + i) * VS_STRIDE + dt1 * 16];
            }
            v16bf a  = cat16(a0, a1);
            v16bf bA = cat16(b0, b1);
            v16bf bB = cat16(c0, c1);
            accA = __builtin_amdgcn_wmma_f32_16x16x32_bf16(false, a, false, bA,
                                                           (short)0, accA, false, false);
            accB = __builtin_amdgcn_wmma_f32_16x16x32_bf16(false, a, false, bB,
                                                           (short)0, accB, false, false);
            if (nt == 0) {   // wave-uniform: EXEC stays all-ones inside
                v8bf s0, s1;
#pragma unroll
                for (int i = 0; i < 8; ++i) {
                    s0[i] = vs[krow + i * VS_STRIDE + 128];
                    s1[i] = vs[krow + (16 + i) * VS_STRIDE + 128];
                }
                accS = __builtin_amdgcn_wmma_f32_16x16x32_bf16(false, a, false, cat16(s0, s1),
                                                               (short)0, accS, false, false);
            }
        }
        __syncthreads();   // protect kt/vs/ps before next chunk's gather
    }

    // ---- finalize: denom = row-sum (col 0 of accS) + exp2(sink - m_final) ----
    if (nt == 0 && l15 == 0) {
#pragma unroll
        for (int r = 0; r < 8; ++r) lsum[mT * 16 + half * 8 + r] = accS[r];
    }
    __syncthreads();
    if (t < NH) rinv[t] = 1.0f / (lsum[t] + EXP2F(sink_s[t] - m_run[t]));
    __syncthreads();

    // ---- store O * (1/denom) ----
    float* op = out + (size_t)s * NH * DD;
#pragma unroll
    for (int r = 0; r < 8; ++r) {
        const int row = mT * 16 + half * 8 + r;
        const float sc_ = rinv[row];
        op[row * DD + dt0 * 16 + l15] = accA[r] * sc_;
        op[row * DD + dt1 * 16 + l15] = accB[r] * sc_;
    }
}

extern "C" void kernel_launch(void* const* d_in, const int* in_sizes, int n_in,
                              void* d_out, int out_size, void* d_ws, size_t ws_size,
                              hipStream_t stream) {
    const float* q    = (const float*)d_in[0];
    const float* kv   = (const float*)d_in[1];
    const float* sink = (const float*)d_in[2];
    const int*   topk = (const int*)d_in[3];
    float* out = (float*)d_out;

    const int Sq = in_sizes[0] / (NH * DD);   // 2048
    sparse_attn_wmma_kernel<<<dim3(Sq), dim3(256), 0, stream>>>(q, kv, sink, topk, out);
}
